// Actor_69887707840990
// MI455X (gfx1250) — compile-verified, matrix operand order checked by reading, not run on hardware
//
#include <hip/hip_runtime.h>
#include <hip/hip_bf16.h>

typedef __attribute__((ext_vector_type(16))) _Float16 v16h;
typedef __attribute__((ext_vector_type(8)))  _Float16 v8h;
typedef __attribute__((ext_vector_type(8)))  float    v8f;

#define BATCH_N 65536
#define ROWS_PER_WAVE 16
#define WAVES_PER_BLK 8
#define ROWS_PER_BLK 128

// LDS layout, offsets in _Float16 units (total 162568 halves = 325,136 B <= 320 KB)
#define OFF_W1T 0            // W1^T  [256][160] f16 (K padded 146->160, zero fill)
#define OFF_WXT 40960        // W2^T then rho_w1^T [256][256] f16
#define OFF_INP 106496       // per-wave input tiles [16][160] f16 x 8
#define OFF_H   126976       // per-wave activation tiles [16][256] f16 x 8
#define OFF_B1  159744       // phi_b1 f16[256]
#define OFF_B2  160000       // phi_b2 f16[256]
#define OFF_RB  160256       // rho_b1 f16[256]
#define OFF_HW  160512       // head weights TRANSPOSED [8][256] f16 (rows 0-3 mean, 4-7 lstd)
#define OFF_HB  162560       // head bias f16[8]
#define SMEM_HALVES 162568

// Load one 16-bit 16x32 WMMA A/B fragment (v16h == 8 VGPRs) from an LDS tile
// stored with contiguous K. Caller passes  base + row*stride + kbase + (lane>=16 ? 8 : 0).
// ISA layout: elems 0..7 = K(kbase + half*8 .. +7), elems 8..15 = K(kbase+16+half*8 ..).
__device__ __forceinline__ v16h load_frag16(const _Float16* p0) {
  v8h lo = *(const v8h*)(p0);
  v8h hi = *(const v8h*)(p0 + 16);
  v16h r;
#pragma unroll
  for (int i = 0; i < 8; ++i) { r[i] = lo[i]; r[i + 8] = hi[i]; }
  return r;
}

__global__ void __launch_bounds__(256)
actor_fused(const float* __restrict__ obs, const float* __restrict__ lemb,
            const float* __restrict__ w1,  const float* __restrict__ b1,
            const float* __restrict__ w2,  const float* __restrict__ b2,
            const float* __restrict__ rw,  const float* __restrict__ rb,
            const float* __restrict__ mw,  const float* __restrict__ mb,
            const float* __restrict__ lw,  const float* __restrict__ lb,
            float* __restrict__ out)
{
  extern __shared__ _Float16 smem[];
  const int tid   = threadIdx.x;
  const int lane  = tid & 31;
  const int wave  = tid >> 5;
  const int lhalf = lane >> 4;        // 0 for lanes 0-15, 1 for lanes 16-31
  const int l15   = lane & 15;
  const int b0w   = blockIdx.x * ROWS_PER_BLK + wave * ROWS_PER_WAVE;

  // Warm L2 with rho_w1 now (used after the pair loop): one prefetch per 128B line.
  for (int idx = tid; idx < 2048; idx += 256)
    __builtin_prefetch(rw + (size_t)idx * 32, 0, 1);

  // ---- cooperative weight staging (f32 global -> f16 transposed LDS) ----
  for (int idx = tid; idx < 160 * 256; idx += 256) {          // W1^T [n][k]
    int k = idx >> 8, n = idx & 255;
    float v = (k < 146) ? w1[k * 256 + n] : 0.0f;
    smem[OFF_W1T + n * 160 + k] = (_Float16)v;
  }
  for (int idx = tid; idx < 256 * 256; idx += 256) {          // W2^T [n][k]
    int k = idx >> 8, n = idx & 255;
    smem[OFF_WXT + n * 256 + k] = (_Float16)w2[k * 256 + n];
  }
  {
    int n = tid;                                              // 256 threads, one each
    smem[OFF_B1 + n] = (_Float16)b1[n];
    smem[OFF_B2 + n] = (_Float16)b2[n];
    smem[OFF_RB + n] = (_Float16)rb[n];
  }
  for (int idx = tid; idx < 2048; idx += 256) {               // heads transposed [c][k]
    int c = idx >> 8, k = idx & 255;
    float v = (c < 4) ? mw[k * 4 + c] : lw[k * 4 + (c - 4)];
    smem[OFF_HW + idx] = (_Float16)v;                         // idx == c*256 + k
  }
  if (tid < 8) smem[OFF_HB + tid] = (_Float16)((tid < 4) ? mb[tid] : lb[tid - 4]);

  // ---- per-wave: base part of the input tile [16][160] (cols 110..145 are per-pair) ----
  _Float16* inp   = smem + OFF_INP + wave * (16 * 160);
  _Float16* htile = smem + OFF_H   + wave * (16 * 256);
  for (int idx = lane; idx < 16 * 160; idx += 32) {
    int r = idx & 15, c = idx >> 4;
    float v;
    if (c < 100)      v = lemb[(size_t)(b0w + r) * 100 + c];
    else if (c < 110) v = obs[(size_t)(b0w + r) * 55 + (c - 100)];
    else if (c < 146) continue;                               // pair-specific region
    else              v = 0.0f;                               // K padding 146..159
    inp[r * 160 + c] = (_Float16)v;
  }
  __syncthreads();

  // agg accumulator: 16x256 tile kept packed f16 in C-layout (64 VGPRs)
  v8h agg[16];
#pragma unroll
  for (int t = 0; t < 16; ++t)
#pragma unroll
    for (int i = 0; i < 8; ++i) agg[t][i] = (_Float16)0.0f;

#pragma unroll 1
  for (int p = 0; p < 6; ++p) {
    const int ii = p >> 1;                                    // IDX_I = {0,0,1,1,2,2}
    const int jj = (0x102021 >> (p * 4)) & 0xF;               // IDX_J = {1,2,0,2,0,1}

    // pair-specific columns 110..145: [onehot_i(3) | obj_i(15) | onehot_j(3) | obj_j(15)]
    for (int idx = lane; idx < 16 * 36; idx += 32) {          // 18 iters, no divergence
      int r = idx & 15, c = idx >> 4;
      float v;
      if (c < 3)       v = (c == ii) ? 1.0f : 0.0f;
      else if (c < 18) v = obs[(size_t)(b0w + r) * 55 + 10 + ii * 15 + (c - 3)];
      else if (c < 21) v = ((c - 18) == jj) ? 1.0f : 0.0f;
      else             v = obs[(size_t)(b0w + r) * 55 + 10 + jj * 15 + (c - 21)];
      inp[r * 160 + (110 + c)] = (_Float16)v;
    }

    // ---- layer 1: h1 = relu(inp @ W1 + b1), 16x160 @ 160x256 ----
    v16h a1[5];
#pragma unroll
    for (int ks = 0; ks < 5; ++ks)
      a1[ks] = load_frag16(inp + l15 * 160 + ks * 32 + (lhalf << 3));

#pragma unroll 1
    for (int nt = 0; nt < 16; ++nt) {
      v8f c1;
#pragma unroll
      for (int i = 0; i < 8; ++i) c1[i] = 0.0f;
#pragma unroll
      for (int ks = 0; ks < 5; ++ks) {
        v16h bf = load_frag16(smem + OFF_W1T + (nt * 16 + l15) * 160 + ks * 32 + (lhalf << 3));
        c1 = __builtin_amdgcn_wmma_f32_16x16x32_f16(false, a1[ks], false, bf,
                                                    (short)0, c1, false, false);
      }
      float bias = (float)smem[OFF_B1 + nt * 16 + l15];
#pragma unroll
      for (int i = 0; i < 8; ++i) {                           // C layout: M=i+8*lhalf, N=l15
        float v = c1[i] + bias;
        v = v > 0.0f ? v : 0.0f;
        htile[(i + 8 * lhalf) * 256 + nt * 16 + l15] = (_Float16)v;
      }
    }

    // ---- layer 2: h2 = relu(h1 @ W2 + b2); agg += h2 ----
    v16h a2[8];
#pragma unroll
    for (int kt = 0; kt < 8; ++kt)
      a2[kt] = load_frag16(htile + l15 * 256 + kt * 32 + (lhalf << 3));

#pragma unroll 1
    for (int nt = 0; nt < 16; ++nt) {
      v8f c2;
#pragma unroll
      for (int i = 0; i < 8; ++i) c2[i] = 0.0f;
#pragma unroll
      for (int kt = 0; kt < 8; ++kt) {
        v16h bf = load_frag16(smem + OFF_WXT + (nt * 16 + l15) * 256 + kt * 32 + (lhalf << 3));
        c2 = __builtin_amdgcn_wmma_f32_16x16x32_f16(false, a2[kt], false, bf,
                                                    (short)0, c2, false, false);
      }
      float bias = (float)smem[OFF_B2 + nt * 16 + l15];
#pragma unroll
      for (int i = 0; i < 8; ++i) {
        float v = c2[i] + bias;
        v = v > 0.0f ? v : 0.0f;
        agg[nt][i] = agg[nt][i] + (_Float16)v;
      }
    }
  }

  // ---- swap W2^T -> rho_w1^T in the shared LDS slot ----
  __syncthreads();                                            // everyone done with W2
  for (int idx = tid; idx < 256 * 256; idx += 256) {
    int k = idx >> 8, n = idx & 255;
    smem[OFF_WXT + n * 256 + k] = (_Float16)rw[k * 256 + n];
  }
  __syncthreads();

  // ---- rho layer: r = relu(agg @ rho_w1 + rho_b1) ----
#pragma unroll
  for (int nt = 0; nt < 16; ++nt)                             // agg -> LDS tile
#pragma unroll
    for (int i = 0; i < 8; ++i)
      htile[(i + 8 * lhalf) * 256 + nt * 16 + l15] = agg[nt][i];

  v16h a3[8];
#pragma unroll
  for (int kt = 0; kt < 8; ++kt)
    a3[kt] = load_frag16(htile + l15 * 256 + kt * 32 + (lhalf << 3));

#pragma unroll 1
  for (int nt = 0; nt < 16; ++nt) {
    v8f c3;
#pragma unroll
    for (int i = 0; i < 8; ++i) c3[i] = 0.0f;
#pragma unroll
    for (int kt = 0; kt < 8; ++kt) {
      v16h bf = load_frag16(smem + OFF_WXT + (nt * 16 + l15) * 256 + kt * 32 + (lhalf << 3));
      c3 = __builtin_amdgcn_wmma_f32_16x16x32_f16(false, a3[kt], false, bf,
                                                  (short)0, c3, false, false);
    }
    float bias = (float)smem[OFF_RB + nt * 16 + l15];
#pragma unroll
    for (int i = 0; i < 8; ++i) {
      float v = c3[i] + bias;
      v = v > 0.0f ? v : 0.0f;
      htile[(i + 8 * lhalf) * 256 + nt * 16 + l15] = (_Float16)v;  // r overwrites agg tile
    }
  }

  // ---- heads: mean = r @ mean_w + mean_b ; log_std = clip(r @ lstd_w + lstd_b) ----
  // Each lane: one r-row (lane>>1), four outputs c = (lane&1)*4 + 0..3.
  // Vector K: v8h loads for r and transposed head weights; 4 independent FMA chains.
  {
    const int row = lane >> 1;
    const int cb  = (lane & 1) * 4;
    float acc[4];
#pragma unroll
    for (int q = 0; q < 4; ++q) acc[q] = (float)smem[OFF_HB + cb + q];

#pragma unroll 1
    for (int k8 = 0; k8 < 32; ++k8) {
      v8h rv = *(const v8h*)(htile + row * 256 + k8 * 8);
#pragma unroll
      for (int q = 0; q < 4; ++q) {
        v8h wv = *(const v8h*)(smem + OFF_HW + (cb + q) * 256 + k8 * 8);
#pragma unroll
        for (int e = 0; e < 8; ++e)
          acc[q] += (float)rv[e] * (float)wv[e];
      }
    }

    const size_t gb = (size_t)(b0w + row);
#pragma unroll
    for (int q = 0; q < 4; ++q) {
      int c = cb + q;
      float a = acc[q];
      if (c < 4) {
        out[gb * 4 + c] = a;
      } else {
        a = a < -20.0f ? -20.0f : (a > 2.0f ? 2.0f : a);
        out[(size_t)BATCH_N * 4 + gb * 4 + (c - 4)] = a;
      }
    }
  }
}

extern "C" void kernel_launch(void* const* d_in, const int* in_sizes, int n_in,
                              void* d_out, int out_size, void* d_ws, size_t ws_size,
                              hipStream_t stream) {
  const float* obs  = (const float*)d_in[0];
  const float* lemb = (const float*)d_in[1];
  const float* w1   = (const float*)d_in[2];
  const float* b1   = (const float*)d_in[3];
  const float* w2   = (const float*)d_in[4];
  const float* b2   = (const float*)d_in[5];
  const float* rw   = (const float*)d_in[6];
  const float* rb   = (const float*)d_in[7];
  const float* mw   = (const float*)d_in[8];
  const float* mb   = (const float*)d_in[9];
  const float* lw   = (const float*)d_in[10];
  const float* lb   = (const float*)d_in[11];
  float* out = (float*)d_out;

  const size_t smem_bytes = (size_t)SMEM_HALVES * 2;          // 325,136 B
  (void)hipFuncSetAttribute((const void*)actor_fused,
                            hipFuncAttributeMaxDynamicSharedMemorySize,
                            (int)smem_bytes);

  dim3 grid(BATCH_N / ROWS_PER_BLK);                          // 512 blocks
  dim3 block(256);                                            // 8 wave32
  hipLaunchKernelGGL(actor_fused, grid, block, smem_bytes, stream,
                     obs, lemb, w1, b1, w2, b2, rw, rb, mw, mb, lw, lb, out);
}
